// MultiLatentAttention_37065567764676
// MI455X (gfx1250) — compile-verified, matrix operand order checked by reading, not run on hardware
//
#include <hip/hip_runtime.h>
#include <hip/hip_bf16.h>
#include <stdint.h>

// ---------------------------------------------------------------------------
// MLA forward for MI455X (gfx1250, wave32).
// - All GEMMs: 3-term split-bf16 (hi/lo) WMMA 16x16x32, f32 accumulation.
// - Operands pre-converted once to bf16 hi/lo (weights pre-transposed).
// - Hot loops stage LDS tiles with global_load_async_to_lds_b128, double-
//   buffered using in-order ASYNCcnt threshold waits.
// ---------------------------------------------------------------------------

typedef __attribute__((ext_vector_type(16))) __bf16 v16bf;
typedef __attribute__((ext_vector_type(8)))  float  v8f;

#define S_LEN   2048
#define NHEADS  16
#define DQK     192
#define DV      128

__device__ __forceinline__ v8f wmma_bf(v16bf a, v16bf b, v8f c) {
  return __builtin_amdgcn_wmma_f32_16x16x32_bf16(false, a, false, b,
                                                 (short)0, c, false, false);
}

// A/B fragment from an LDS tile (row-major bf16). 16-bit 16x32 layout:
// M(or N) = lane&15, kbase = (lane>>4)*8; elems 0..7 -> K=kbase.., 8..15 -> +16.
__device__ __forceinline__ v16bf ld_frag(const __bf16* base, int row,
                                         int stride, int kbase) {
  const __bf16* p = base + row * stride + kbase;
  union { v16bf v; uint4 u[2]; } f;
  f.u[0] = *(const uint4*)(p);
  f.u[1] = *(const uint4*)(p + 16);
  return f.v;
}

// Same fragment pattern loaded straight from global (row pointer pre-offset).
__device__ __forceinline__ v16bf ld_frag_g(const __bf16* rowp, int kbase) {
  union { v16bf v; uint4 u[2]; } f;
  f.u[0] = *(const uint4*)(rowp + kbase);
  f.u[1] = *(const uint4*)(rowp + kbase + 16);
  return f.v;
}

// CDNA5 async global->LDS copy, 16 bytes per lane (tracked by ASYNCcnt).
__device__ __forceinline__ void async_b128(void* lds, const void* g) {
  uint32_t l = (uint32_t)(uintptr_t)lds;   // LDS aperture: addr[31:0] = offset
  asm volatile("global_load_async_to_lds_b128 %0, %1, off"
               :: "v"(l), "v"(g) : "memory");
}
__device__ __forceinline__ void async_wait0() {
  asm volatile("s_wait_asynccnt 0x0" ::: "memory");
}
__device__ __forceinline__ void async_wait4() {
  asm volatile("s_wait_asynccnt 0x4" ::: "memory");
}
__device__ __forceinline__ void async_wait6() {
  asm volatile("s_wait_asynccnt 0x6" ::: "memory");
}

// ---------------------------------------------------------------------------
// Split-bf16 GEMM: C[M,N] = A[M,K] * B[N,K]^T, operands bf16 hi/lo.
// 128x64 macro-tile, 256 threads (8 waves), wave computes 32x32, K-step 32.
// Double-buffered async staging: issue next tile, wait<=6 (in-order), compute.
// ---------------------------------------------------------------------------
#define GSTR 40
#define ABUF (128 * GSTR)
#define BBUF (64 * GSTR)

__global__ __launch_bounds__(256)
void gemm_bf16s_kernel(const __bf16* __restrict__ Ahg, const __bf16* __restrict__ Alg,
                       const __bf16* __restrict__ Bhg, const __bf16* __restrict__ Blg,
                       float* __restrict__ C, int M, int N, int K) {
  __shared__ __align__(16) __bf16 Ahs[2 * ABUF];
  __shared__ __align__(16) __bf16 Als[2 * ABUF];
  __shared__ __align__(16) __bf16 Bhs[2 * BBUF];
  __shared__ __align__(16) __bf16 Bls[2 * BBUF];

  const int tid  = threadIdx.x;
  const int lane = tid & 31;
  const int wave = tid >> 5;
  const int l15  = lane & 15;
  const int lh   = lane >> 4;
  const int wr   = wave >> 1;   // 0..3 -> 32-row strip
  const int wc   = wave & 1;    // 0..1 -> 32-col strip
  const int m0 = blockIdx.y * 128;
  const int n0 = blockIdx.x * 64;

  v8f zero = {0.f,0.f,0.f,0.f,0.f,0.f,0.f,0.f};
  v8f acc00 = zero, acc01 = zero, acc10 = zero, acc11 = zero;

  const int ar = tid >> 1;           // 0..127
  const int ac = (tid & 1) * 16;     // 0 or 16
  const int bn = tid >> 2;           // 0..63
  const int bc = (tid & 3) * 8;      // 0..24

  auto issue = [&](int buf, int k0) {
    size_t ga = (size_t)(m0 + ar) * K + k0 + ac;
    async_b128(&Ahs[buf * ABUF + ar * GSTR + ac],     Ahg + ga);
    async_b128(&Ahs[buf * ABUF + ar * GSTR + ac + 8], Ahg + ga + 8);
    async_b128(&Als[buf * ABUF + ar * GSTR + ac],     Alg + ga);
    async_b128(&Als[buf * ABUF + ar * GSTR + ac + 8], Alg + ga + 8);
    size_t gb = (size_t)(n0 + bn) * K + k0 + bc;
    async_b128(&Bhs[buf * BBUF + bn * GSTR + bc], Bhg + gb);
    async_b128(&Bls[buf * BBUF + bn * GSTR + bc], Blg + gb);
  };

  const int nsteps = K >> 5;
  issue(0, 0);

  for (int step = 0; step < nsteps; ++step) {
    const int cur = step & 1;
    if (step + 1 < nsteps) {
      issue(cur ^ 1, (step + 1) * 32);
      async_wait6();                 // in-order: current tile's 6 copies done
    } else {
      async_wait0();
    }
    __syncthreads();

    const __bf16* Ah = &Ahs[cur * ABUF];
    const __bf16* Al = &Als[cur * ABUF];
    const __bf16* Bh = &Bhs[cur * BBUF];
    const __bf16* Bl = &Bls[cur * BBUF];
    const int kbse = lh * 8;
    v16bf a0h = ld_frag(Ah, wr * 32 + l15,      GSTR, kbse);
    v16bf a0l = ld_frag(Al, wr * 32 + l15,      GSTR, kbse);
    v16bf a1h = ld_frag(Ah, wr * 32 + 16 + l15, GSTR, kbse);
    v16bf a1l = ld_frag(Al, wr * 32 + 16 + l15, GSTR, kbse);
    v16bf b0h = ld_frag(Bh, wc * 32 + l15,      GSTR, kbse);
    v16bf b0l = ld_frag(Bl, wc * 32 + l15,      GSTR, kbse);
    v16bf b1h = ld_frag(Bh, wc * 32 + 16 + l15, GSTR, kbse);
    v16bf b1l = ld_frag(Bl, wc * 32 + 16 + l15, GSTR, kbse);

    acc00 = wmma_bf(a0h, b0h, acc00);
    acc00 = wmma_bf(a0h, b0l, acc00);
    acc00 = wmma_bf(a0l, b0h, acc00);
    acc01 = wmma_bf(a0h, b1h, acc01);
    acc01 = wmma_bf(a0h, b1l, acc01);
    acc01 = wmma_bf(a0l, b1h, acc01);
    acc10 = wmma_bf(a1h, b0h, acc10);
    acc10 = wmma_bf(a1h, b0l, acc10);
    acc10 = wmma_bf(a1l, b0h, acc10);
    acc11 = wmma_bf(a1h, b1h, acc11);
    acc11 = wmma_bf(a1h, b1l, acc11);
    acc11 = wmma_bf(a1l, b1h, acc11);
    __syncthreads();                 // reads done before buffer is re-filled
  }

#pragma unroll
  for (int r = 0; r < 8; ++r) {
    int row = m0 + wr * 32 + r + 8 * lh;
    int col = n0 + wc * 32 + l15;
    C[(size_t)row * N + col]             = acc00[r];
    C[(size_t)row * N + col + 16]        = acc01[r];
    C[(size_t)(row + 16) * N + col]      = acc10[r];
    C[(size_t)(row + 16) * N + col + 16] = acc11[r];
  }
}

// ---------------------------------------------------------------------------
// fp32 -> split bf16 (elementwise)
// ---------------------------------------------------------------------------
__global__ __launch_bounds__(256)
void split_kernel(const float* __restrict__ x, __bf16* __restrict__ yh,
                  __bf16* __restrict__ yl, int n) {
  int i = blockIdx.x * 256 + threadIdx.x;
  if (i < n) {
    float v = x[i];
    __bf16 h = (__bf16)v;
    yh[i] = h;
    yl[i] = (__bf16)(v - (float)h);
  }
}

// W[K][N] f32 -> Wt_hi/Wt_lo [N][K] bf16 (tiled transpose + split)
__global__ __launch_bounds__(256)
void transpose_split_kernel(const float* __restrict__ W, __bf16* __restrict__ Th,
                            __bf16* __restrict__ Tl, int K, int N) {
  __shared__ float tile[32][33];
  const int tx = threadIdx.x & 31, ty = threadIdx.x >> 5;  // 32 x 8
  const int kb = blockIdx.y * 32, nb = blockIdx.x * 32;
#pragma unroll
  for (int i = 0; i < 32; i += 8)
    tile[ty + i][tx] = W[(size_t)(kb + ty + i) * N + nb + tx];
  __syncthreads();
#pragma unroll
  for (int i = 0; i < 32; i += 8) {
    float v = tile[tx][ty + i];
    __bf16 h = (__bf16)v;
    size_t o = (size_t)(nb + ty + i) * K + kb + tx;
    Th[o] = h;
    Tl[o] = (__bf16)(v - (float)h);
  }
}

// ---------------------------------------------------------------------------
// RMSNorm -> split bf16 output
// ---------------------------------------------------------------------------
__global__ __launch_bounds__(256)
void rmsnorm_kernel(const float* __restrict__ x, const float* __restrict__ g,
                    __bf16* __restrict__ yh, __bf16* __restrict__ yl,
                    int L, int istride) {
  __shared__ float red[256];
  const int row = blockIdx.x;
  const int tid = threadIdx.x;
  float s = 0.f;
  for (int i = tid; i < L; i += 256) {
    float v = x[(size_t)row * istride + i];
    s += v * v;
  }
  red[tid] = s;
  __syncthreads();
  for (int off = 128; off > 0; off >>= 1) {
    if (tid < off) red[tid] += red[tid + off];
    __syncthreads();
  }
  float scale = rsqrtf(red[0] / (float)L + 1e-6f);
  for (int i = tid; i < L; i += 256) {
    float v = x[(size_t)row * istride + i] * scale * g[i];
    __bf16 h = (__bf16)v;
    yh[(size_t)row * L + i] = h;
    yl[(size_t)row * L + i] = (__bf16)(v - (float)h);
  }
}

// ---------------------------------------------------------------------------
// RoPE (deinterleave variant) + per-head split-bf16 layout.
// ---------------------------------------------------------------------------
__device__ __forceinline__ float rope_elem(const float* pe, const float* cosb,
                                           const float* sinb, int s, int j) {
  float c  = cosb[s * 64 + j];
  float sn = sinb[s * 64 + j];
  if (j < 32) return pe[2 * j] * c - pe[2 * j + 1] * sn;
  int i = j - 32;
  return pe[2 * i + 1] * c + pe[2 * i] * sn;
}

__global__ __launch_bounds__(256)
void assemble_q_kernel(const float* __restrict__ q, const float* __restrict__ cosb,
                       const float* __restrict__ sinb,
                       __bf16* __restrict__ Qh, __bf16* __restrict__ Ql) {
  const int s = blockIdx.x;
  const float sc = 0.0721687836487032f;  // 1/sqrt(192), folded into Q
  for (int e = threadIdx.x; e < NHEADS * DQK; e += 256) {
    int hh = e / DQK, d = e - hh * DQK;
    const float* qrow = q + (size_t)s * (NHEADS * DQK) + hh * DQK;
    float v = (d < 128) ? qrow[d] : rope_elem(qrow + 128, cosb, sinb, s, d - 128);
    v *= sc;
    __bf16 h = (__bf16)v;
    size_t o = ((size_t)hh * S_LEN + s) * DQK + d;
    Qh[o] = h;
    Ql[o] = (__bf16)(v - (float)h);
  }
}

__global__ __launch_bounds__(256)
void assemble_k_kernel(const float* __restrict__ kv, const float* __restrict__ ckv,
                       const float* __restrict__ cosb, const float* __restrict__ sinb,
                       __bf16* __restrict__ Kh, __bf16* __restrict__ Kl) {
  const int s = blockIdx.x;
  for (int e = threadIdx.x; e < NHEADS * DQK; e += 256) {
    int hh = e / DQK, d = e - hh * DQK;
    float v;
    if (d < 128) v = kv[(size_t)s * 4096 + hh * 256 + d];
    else         v = rope_elem(ckv + (size_t)s * 576 + 512, cosb, sinb, s, d - 128);
    __bf16 h = (__bf16)v;
    size_t o = ((size_t)hh * S_LEN + s) * DQK + d;
    Kh[o] = h;
    Kl[o] = (__bf16)(v - (float)h);
  }
}

// V -> bf16, pre-transposed to [h][d][s] so flash tiles are contiguous copies.
__global__ __launch_bounds__(256)
void assemble_v_kernel(const float* __restrict__ kv, __bf16* __restrict__ Vt) {
  const int hd = blockIdx.x;                 // 0 .. 16*128-1
  const int hh = hd >> 7, d = hd & 127;
  for (int s = threadIdx.x; s < S_LEN; s += 256) {
    Vt[(size_t)hd * S_LEN + s] = (__bf16)kv[(size_t)s * 4096 + hh * 256 + 128 + d];
  }
}

// ---------------------------------------------------------------------------
// Causal flash attention: grid (S/64, NH), 128 threads (4 waves x 16 q-rows).
// Q fragments resident in registers (loaded once from global); K chunks
// double-buffered via async LDS; V tile async; bf16 P*V.
// ---------------------------------------------------------------------------
#define KSTR 40
#define KBUF (64 * KSTR)
#define VSTR 72
#define PSTR 72

__global__ __launch_bounds__(128)
void flash_kernel(const __bf16* __restrict__ Qhg, const __bf16* __restrict__ Qlg,
                  const __bf16* __restrict__ Khg, const __bf16* __restrict__ Klg,
                  const __bf16* __restrict__ Vtg,
                  __bf16* __restrict__ Oh, __bf16* __restrict__ Ol) {
  __shared__ __align__(16) __bf16 Kch[2 * KBUF];
  __shared__ __align__(16) __bf16 Kcl[2 * KBUF];
  __shared__ __align__(16) __bf16 Vt[128 * VSTR];   // [d][key]
  __shared__ __align__(16) __bf16 P[64 * PSTR];

  const int tid  = threadIdx.x;
  const int lane = tid & 31;
  const int wave = tid >> 5;
  const int l15  = lane & 15;
  const int lh   = lane >> 4;
  const int q0   = blockIdx.x * 64;
  const int h    = blockIdx.y;

  // Q fragments in registers: 6 d-chunks x (hi, lo)
  v16bf qfh[6], qfl[6];
  {
    size_t rowoff = ((size_t)h * S_LEN + q0 + wave * 16 + l15) * DQK;
    const __bf16* qrh = Qhg + rowoff;
    const __bf16* qrl = Qlg + rowoff;
#pragma unroll
    for (int kc = 0; kc < 6; ++kc) {
      int kbase = kc * 32 + lh * 8;
      qfh[kc] = ld_frag_g(qrh, kbase);
      qfl[kc] = ld_frag_g(qrl, kbase);
    }
  }

  v8f zero = {0.f,0.f,0.f,0.f,0.f,0.f,0.f,0.f};
  v8f Oacc[8];
#pragma unroll
  for (int i = 0; i < 8; ++i) Oacc[i] = zero;
  float mrow[8], lrow[8];
#pragma unroll
  for (int r = 0; r < 8; ++r) { mrow[r] = -1e30f; lrow[r] = 0.f; }

  const int cr = tid >> 1;          // 0..63 (K chunk stage row)
  const int cc = (tid & 1) * 16;    // 0/16

  auto issueK = [&](int buf, int kb, int kc) {
    size_t gk = ((size_t)h * S_LEN + kb * 64 + cr) * DQK + kc * 32 + cc;
    async_b128(&Kch[buf * KBUF + cr * KSTR + cc],     Khg + gk);
    async_b128(&Kch[buf * KBUF + cr * KSTR + cc + 8], Khg + gk + 8);
    async_b128(&Kcl[buf * KBUF + cr * KSTR + cc],     Klg + gk);
    async_b128(&Kcl[buf * KBUF + cr * KSTR + cc + 8], Klg + gk + 8);
  };

  const int nkb = q0 / 64 + 1;
  for (int kb = 0; kb < nkb; ++kb) {
    __syncthreads();  // previous iteration's Vt/P/Kbuf reads complete
    {   // V tile: Vt[d][k] <- Vtg[h][d][kb*64+k], one d-row per thread
      const __bf16* src = Vtg + ((size_t)h * DV + tid) * S_LEN + kb * 64;
#pragma unroll
      for (int j = 0; j < 8; ++j)
        async_b128(&Vt[tid * VSTR + j * 8], src + j * 8);
    }
    issueK(0, kb, 0);

    v8f Sf[4];
#pragma unroll
    for (int cf = 0; cf < 4; ++cf) Sf[cf] = zero;

#pragma unroll
    for (int kc = 0; kc < 6; ++kc) {  // d = 192 in chunks of 32
      const int cur = kc & 1;
      if (kc < 5) {
        issueK(cur ^ 1, kb, kc + 1);
        async_wait4();                // in-order: chunk kc (and V) complete
      } else {
        async_wait0();
      }
      __syncthreads();

      const __bf16* Kh = &Kch[cur * KBUF];
      const __bf16* Kl = &Kcl[cur * KBUF];
      const int kbse = lh * 8;
#pragma unroll
      for (int cf = 0; cf < 4; ++cf) {
        v16bf kh = ld_frag(Kh, cf * 16 + l15, KSTR, kbse);
        v16bf kl = ld_frag(Kl, cf * 16 + l15, KSTR, kbse);
        Sf[cf] = wmma_bf(qfh[kc], kh, Sf[cf]);
        Sf[cf] = wmma_bf(qfh[kc], kl, Sf[cf]);
        Sf[cf] = wmma_bf(qfl[kc], kh, Sf[cf]);
      }
      __syncthreads();  // reads done before this buffer is re-filled
    }

    if (kb == nkb - 1) {  // causal mask on diagonal block
#pragma unroll
      for (int cf = 0; cf < 4; ++cf) {
        int key = kb * 64 + cf * 16 + l15;
#pragma unroll
        for (int r = 0; r < 8; ++r) {
          int qr = q0 + wave * 16 + r + 8 * lh;
          if (key > qr) Sf[cf][r] = -1e30f;
        }
      }
    }

    // Online softmax; C-frag slot r <-> row r + 8*lh within each 16-lane half.
    float mnew[8];
#pragma unroll
    for (int r = 0; r < 8; ++r) {
      float mx = fmaxf(fmaxf(Sf[0][r], Sf[1][r]), fmaxf(Sf[2][r], Sf[3][r]));
      mx = fmaxf(mx, __shfl_xor(mx, 1, 32));
      mx = fmaxf(mx, __shfl_xor(mx, 2, 32));
      mx = fmaxf(mx, __shfl_xor(mx, 4, 32));
      mx = fmaxf(mx, __shfl_xor(mx, 8, 32));
      mnew[r] = fmaxf(mrow[r], mx);
    }
#pragma unroll
    for (int r = 0; r < 8; ++r) {
      float alpha = __expf(mrow[r] - mnew[r]);
      lrow[r] *= alpha;
#pragma unroll
      for (int nf = 0; nf < 8; ++nf) Oacc[nf][r] *= alpha;
      mrow[r] = mnew[r];
    }
#pragma unroll
    for (int r = 0; r < 8; ++r) {
      float rs = 0.f;
#pragma unroll
      for (int cf = 0; cf < 4; ++cf) {
        float p = __expf(Sf[cf][r] - mnew[r]);
        rs += p;
        P[(wave * 16 + r + 8 * lh) * PSTR + cf * 16 + l15] = (__bf16)p;
      }
      rs += __shfl_xor(rs, 1, 32);
      rs += __shfl_xor(rs, 2, 32);
      rs += __shfl_xor(rs, 4, 32);
      rs += __shfl_xor(rs, 8, 32);
      lrow[r] += rs;
    }
    __syncthreads();  // P C-layout write -> A-frag cross-lane reads

    // O += P(16x64) * V(64x128)
#pragma unroll
    for (int kc2 = 0; kc2 < 2; ++kc2) {
      const int kbse = kc2 * 32 + lh * 8;
      v16bf pf = ld_frag(P, wave * 16 + l15, PSTR, kbse);
#pragma unroll
      for (int nf = 0; nf < 8; ++nf) {
        v16bf vf = ld_frag(Vt, nf * 16 + l15, VSTR, kbse);
        Oacc[nf] = wmma_bf(pf, vf, Oacc[nf]);
      }
    }
  }

  // Normalize; emit split-bf16 attn rows [s][h*128+d] for the output GEMM.
#pragma unroll
  for (int r = 0; r < 8; ++r) {
    float inv = 1.f / lrow[r];
    int row = q0 + wave * 16 + r + 8 * lh;
#pragma unroll
    for (int nf = 0; nf < 8; ++nf) {
      float v = Oacc[nf][r] * inv;
      __bf16 hb = (__bf16)v;
      size_t o = (size_t)row * (NHEADS * DV) + h * DV + nf * 16 + l15;
      Oh[o] = hb;
      Ol[o] = (__bf16)(v - (float)hb);
    }
  }
}

// ---------------------------------------------------------------------------
// Host pipeline.
// ---------------------------------------------------------------------------
extern "C" void kernel_launch(void* const* d_in, const int* in_sizes, int n_in,
                              void* d_out, int out_size, void* d_ws, size_t ws_size,
                              hipStream_t stream) {
  (void)in_sizes; (void)n_in; (void)out_size; (void)ws_size;

  const float* hs    = (const float*)d_in[0];
  const float* cosb  = (const float*)d_in[1];
  const float* sinb  = (const float*)d_in[2];
  const float* w_qa  = (const float*)d_in[3];   // 2048x1536
  const float* g_qa  = (const float*)d_in[4];
  const float* w_qb  = (const float*)d_in[5];   // 1536x3072
  const float* w_kva = (const float*)d_in[6];   // 2048x576
  const float* g_kva = (const float*)d_in[7];
  const float* w_kvb = (const float*)d_in[8];   // 512x4096
  const float* w_o   = (const float*)d_in[9];   // 2048x2048

  char* ws = (char*)d_ws;
  // fresh allocations
  __bf16* hsh    = (__bf16*)(ws + 0);            // 2048*2048
  __bf16* hsl    = (__bf16*)(ws + 8388608);
  __bf16* WqaT_h = (__bf16*)(ws + 16777216);     // [1536][2048]
  __bf16* WqaT_l = (__bf16*)(ws + 23068672);
  float*  q_lat  = (float*) (ws + 29360128);     // 2048x1536
  __bf16* qn_h   = (__bf16*)(ws + 41943040);     // 2048x1536
  __bf16* qn_l   = (__bf16*)(ws + 48234496);
  __bf16* WqbT_h = (__bf16*)(ws + 54525952);     // [3072][1536]
  __bf16* WqbT_l = (__bf16*)(ws + 63963136);
  float*  q      = (float*) (ws + 73400320);     // 2048x3072
  __bf16* WkvaT_h= (__bf16*)(ws + 98566144);     // [576][2048]
  __bf16* WkvaT_l= (__bf16*)(ws + 100925440);
  float*  ckv    = (float*) (ws + 103284736);    // 2048x576
  __bf16* ckvn_h = (__bf16*)(ws + 108003328);    // 2048x512
  __bf16* ckvn_l = (__bf16*)(ws + 110100480);
  __bf16* WkvbT_h= (__bf16*)(ws + 112197632);    // [4096][512]
  __bf16* WkvbT_l= (__bf16*)(ws + 116391936);
  float*  kv     = (float*) (ws + 120586240);    // 2048x4096
  __bf16* WoT_h  = (__bf16*)(ws + 154140672);    // [2048][2048]
  __bf16* WoT_l  = (__bf16*)(ws + 162529280);
  // reused regions (producers run after the original owners are dead)
  __bf16* Qs_h   = (__bf16*)(ws + 0);            // over hs (dead after ckv GEMM)
  __bf16* Qs_l   = (__bf16*)(ws + 12582912);     // spills into dead WqaT
  __bf16* Ks_h   = (__bf16*)(ws + 29360128);     // over q_lat/qn (dead)
  __bf16* Ks_l   = (__bf16*)(ws + 41943040);
  __bf16* Vt_g   = (__bf16*)(ws + 54525952);     // over WqbT (dead)
  __bf16* attn_h = (__bf16*)(ws + 98566144);     // over WkvaT/ckv/ckvn/WkvbT
  __bf16* attn_l = (__bf16*)(ws + 106954752);
  float*  out    = (float*)d_out;

  dim3 blk(256);

  // pre-pass conversions
  split_kernel<<<16384, blk, 0, stream>>>(hs, hsh, hsl, 2048 * 2048);
  transpose_split_kernel<<<dim3(1536/32, 2048/32), blk, 0, stream>>>(w_qa, WqaT_h, WqaT_l, 2048, 1536);
  transpose_split_kernel<<<dim3(3072/32, 1536/32), blk, 0, stream>>>(w_qb, WqbT_h, WqbT_l, 1536, 3072);
  transpose_split_kernel<<<dim3(576/32,  2048/32), blk, 0, stream>>>(w_kva, WkvaT_h, WkvaT_l, 2048, 576);
  transpose_split_kernel<<<dim3(4096/32, 512/32),  blk, 0, stream>>>(w_kvb, WkvbT_h, WkvbT_l, 512, 4096);
  transpose_split_kernel<<<dim3(2048/32, 2048/32), blk, 0, stream>>>(w_o, WoT_h, WoT_l, 2048, 2048);

  // q path
  gemm_bf16s_kernel<<<dim3(1536/64, 2048/128), blk, 0, stream>>>(hsh, hsl, WqaT_h, WqaT_l, q_lat, 2048, 1536, 2048);
  rmsnorm_kernel<<<2048, blk, 0, stream>>>(q_lat, g_qa, qn_h, qn_l, 1536, 1536);
  gemm_bf16s_kernel<<<dim3(3072/64, 2048/128), blk, 0, stream>>>(qn_h, qn_l, WqbT_h, WqbT_l, q, 2048, 3072, 1536);

  // kv path
  gemm_bf16s_kernel<<<dim3(576/64, 2048/128), blk, 0, stream>>>(hsh, hsl, WkvaT_h, WkvaT_l, ckv, 2048, 576, 2048);
  rmsnorm_kernel<<<2048, blk, 0, stream>>>(ckv, g_kva, ckvn_h, ckvn_l, 512, 576);
  gemm_bf16s_kernel<<<dim3(4096/64, 2048/128), blk, 0, stream>>>(ckvn_h, ckvn_l, WkvbT_h, WkvbT_l, kv, 2048, 4096, 512);

  // rope + per-head layout (Qs first: it reads q; Ks/Vt overwrite dead buffers)
  assemble_q_kernel<<<2048, blk, 0, stream>>>(q, cosb, sinb, Qs_h, Qs_l);
  assemble_k_kernel<<<2048, blk, 0, stream>>>(kv, ckv, cosb, sinb, Ks_h, Ks_l);
  assemble_v_kernel<<<NHEADS * DV, blk, 0, stream>>>(kv, Vt_g);

  // causal flash attention -> split-bf16 attn
  flash_kernel<<<dim3(2048/64, NHEADS), dim3(128), 0, stream>>>(
      Qs_h, Qs_l, Ks_h, Ks_l, Vt_g, attn_h, attn_l);

  // output projection
  gemm_bf16s_kernel<<<dim3(2048/64, 2048/128), blk, 0, stream>>>(
      attn_h, attn_l, WoT_h, WoT_l, out, 2048, 2048, 2048);
}
